// SparseConv_45715631898990
// MI455X (gfx1250) — matrix operand order repaired and checked by reading, not compiled, not run on hardware
//
#include <hip/hip_runtime.h>

typedef __attribute__((ext_vector_type(16))) _Float16 v16h;
typedef __attribute__((ext_vector_type(8)))  _Float16 v8h;
typedef __attribute__((ext_vector_type(2)))  _Float16 v2h;
typedef __attribute__((ext_vector_type(8)))  float    v8f;

#define CIN   64
#define COUT  64
#define NBUCK 32768
#define BSH   15

static __device__ __forceinline__ bool any_lane(bool p) {
#if __has_builtin(__builtin_amdgcn_ballot_w32)
  return __builtin_amdgcn_ballot_w32(p) != 0u;   // wave32 ballot
#else
  return __any(p);
#endif
}

// ---- zero bucket histogram + scatter counters ----
__global__ void k_init(int* __restrict__ hist, int* __restrict__ counters) {
  int t = blockIdx.x * blockDim.x + threadIdx.x;
  if (t < NBUCK) { hist[t] = 0; counters[t] = 0; }
}

// ---- pack coords to keys + bucket histogram ----
__global__ void k_keys(const int* __restrict__ coords, int* __restrict__ keys,
                       int* __restrict__ hist, int n) {
  int i = blockIdx.x * blockDim.x + threadIdx.x;
  if (i >= n) return;
  int x = coords[3 * i], y = coords[3 * i + 1], z = coords[3 * i + 2];
  int key = (x * 1024 + y) * 1024 + z;
  keys[i] = key;
  atomicAdd(&hist[key >> BSH], 1);
}

// ---- exclusive scan of 32768 bins, single workgroup ----
__global__ void k_scan(const int* __restrict__ hist, int* __restrict__ starts) {
  __shared__ int part[1024];
  int t = threadIdx.x;
  int s = 0;
  for (int j = 0; j < 32; ++j) s += hist[t * 32 + j];
  part[t] = s;
  __syncthreads();
  for (int off = 1; off < 1024; off <<= 1) {
    int v = (t >= off) ? part[t - off] : 0;
    __syncthreads();
    part[t] += v;
    __syncthreads();
  }
  int run = (t == 0) ? 0 : part[t - 1];
  for (int j = 0; j < 32; ++j) {
    starts[t * 32 + j] = run;
    run += hist[t * 32 + j];
  }
  if (t == 1023) starts[NBUCK] = run;
}

// ---- scatter (key,idx) into bucket regions ----
__global__ void k_scatter(const int* __restrict__ keys, const int* __restrict__ starts,
                          int* __restrict__ counters, int* __restrict__ skeys,
                          int* __restrict__ sidx, int n) {
  int i = blockIdx.x * blockDim.x + threadIdx.x;
  if (i >= n) return;
  int key = keys[i];
  int b = key >> BSH;
  int pos = starts[b] + atomicAdd(&counters[b], 1);
  skeys[pos] = key;
  sidx[pos] = i;
}

// ---- per-bucket insertion sort (avg bucket ~12; keys unique => deterministic) ----
__global__ void k_bsort(const int* __restrict__ starts, int* __restrict__ skeys,
                        int* __restrict__ sidx) {
  int b = blockIdx.x * blockDim.x + threadIdx.x;
  if (b >= NBUCK) return;
  int lo = starts[b], hi = starts[b + 1];
  for (int i = lo + 1; i < hi; ++i) {
    int kv = skeys[i], iv = sidx[i];
    int j = i - 1;
    while (j >= lo && skeys[j] > kv) {
      skeys[j + 1] = skeys[j];
      sidx[j + 1] = sidx[j];
      --j;
    }
    skeys[j + 1] = kv;
    sidx[j + 1] = iv;
  }
}

// ---- gather features into sorted order, f32 -> f16 ----
__global__ void k_gather(const float* __restrict__ feat, const int* __restrict__ sidx,
                         _Float16* __restrict__ sfeat, int n) {
  int t = blockIdx.x * blockDim.x + threadIdx.x;
  if (t >= n * 32) return;
  int r = t >> 5, c2 = (t & 31) * 2;
  const float* fp = feat + (size_t)sidx[r] * CIN + c2;
  v2h h = {};
  h[0] = (_Float16)fp[0];
  h[1] = (_Float16)fp[1];
  *(v2h*)(sfeat + (size_t)r * CIN + c2) = h;
}

// ---- transpose weights [k][cin][cout] f32 -> [k][cout][cin] f16 ----
__global__ void k_prep_w(const float* __restrict__ kern, _Float16* __restrict__ wt, int total) {
  int t = blockIdx.x * blockDim.x + threadIdx.x;
  if (t >= total) return;
  int k = t >> 12;
  int r = t & 4095;
  int co = r >> 6, ci = r & 63;
  wt[((size_t)k * 64 + co) * 64 + ci] = (_Float16)kern[((size_t)k * 64 + ci) * 64 + co];
}

// ---- packed key offsets (linear packing works for negatives) ----
__global__ void k_okeys(const int* __restrict__ offsets, int* __restrict__ okeys, int noff) {
  int k = blockIdx.x * blockDim.x + threadIdx.x;
  if (k < noff)
    okeys[k] = (offsets[3 * k] * 1024 + offsets[3 * k + 1]) * 1024 + offsets[3 * k + 2];
}

// ---- main: one wave32 per 16-row output tile ----
// Per offset: bucketed lower_bound per row; if NO row in the tile matched,
// skip gather+GEMM entirely (exact: missing rows contribute 0*W = 0).
__global__ __launch_bounds__(32) void k_conv(
    const int* __restrict__ skeys, const int* __restrict__ starts,
    const _Float16* __restrict__ sfeat, const _Float16* __restrict__ wt,
    const int* __restrict__ okeys, float* __restrict__ out, int n, int noff) {
  const int l = threadIdx.x;
  const int m = l & 15;                 // row (A) / column (B,D) this lane owns
  const int tile = blockIdx.x;
  const int row = tile * 16 + m;
  const bool rowok = (row < n);
  const int mykey = rowok ? skeys[row] : 0;
  const int hb = (l < 16) ? 0 : 8;      // A fragment K sub-base (halves)
  const int kb = (l < 16) ? 0 : 8;      // B fragment K sub-base (halves)

  v8f acc[4] = {};                      // 4 N-tiles of 16 cols (f32 accum)

  for (int k = 0; k < noff; ++k) {
    // bucketed lower_bound: ~4 probes instead of ~19 over the full array
    const int q = mykey + okeys[k];
    const int b = q >> BSH;
    int lo = starts[b];
    const int hi0 = starts[b + 1];
    int hi = hi0;
    while (lo < hi) {
      int mid = (lo + hi) >> 1;
      if (skeys[mid] < q) lo = mid + 1; else hi = mid;
    }
    const bool valid = rowok && (lo < hi0) && (skeys[lo] == q);

    // Wave-uniform skip: no matches in this tile for this offset -> zero
    // contribution, no gather / no GEMM. (~26/27 of offsets at this density.)
    if (!any_lane(valid)) continue;

    // A fragments (K=0..31 and K=32..63), loaded directly in WMMA layout:
    // lane<16 needs halves {hb..hb+7, hb+16..hb+23} of its gathered row.
    v16h a0 = {}, a1 = {};
    if (valid) {
      const _Float16* rp = sfeat + (size_t)lo * CIN + hb;
      v8h p0 = *(const v8h*)(rp);
      v8h p1 = *(const v8h*)(rp + 16);
      v8h p2 = *(const v8h*)(rp + 32);
      v8h p3 = *(const v8h*)(rp + 48);
      a0 = __builtin_shufflevector(p0, p1, 0,1,2,3,4,5,6,7,8,9,10,11,12,13,14,15);
      a1 = __builtin_shufflevector(p2, p3, 0,1,2,3,4,5,6,7,8,9,10,11,12,13,14,15);
    }

    // Hoist all 8 B fragments (one load clause, one wait), then 8 WMMAs
    // back-to-back. WT[k][co][ci]; b[j] = WT[co][c*32+kb+j] for j<8, WT[co][c*32+kb+8+j] for j>=8.
    const _Float16* wk = wt + (size_t)k * (CIN * COUT) + (size_t)m * CIN + kb;
    v16h bf[8];
#pragma unroll
    for (int nt = 0; nt < 4; ++nt) {
      const _Float16* wp = wk + nt * 16 * CIN;
      v8h q0 = *(const v8h*)(wp);
      v8h q1 = *(const v8h*)(wp + 16);
      v8h q2 = *(const v8h*)(wp + 32);
      v8h q3 = *(const v8h*)(wp + 48);
      bf[2 * nt]     = __builtin_shufflevector(q0, q1, 0,1,2,3,4,5,6,7,8,9,10,11,12,13,14,15);
      bf[2 * nt + 1] = __builtin_shufflevector(q2, q3, 0,1,2,3,4,5,6,7,8,9,10,11,12,13,14,15);
    }
#pragma unroll
    for (int nt = 0; nt < 4; ++nt) {
      acc[nt] = __builtin_amdgcn_wmma_f32_16x16x32_f16(
          false, a0, false, bf[2 * nt], (short)0, acc[nt], false, false);
      acc[nt] = __builtin_amdgcn_wmma_f32_16x16x32_f16(
          false, a1, false, bf[2 * nt + 1], (short)0, acc[nt], false, false);
    }
  }

  // D layout: VGPR j -> row M=j (lanes 0-15) or M=j+8 (lanes 16-31), col N=m
  const int rbase = tile * 16 + ((l < 16) ? 0 : 8);
#pragma unroll
  for (int nt = 0; nt < 4; ++nt) {
#pragma unroll
    for (int j = 0; j < 8; ++j) {
      int r = rbase + j;
      if (r < n) out[(size_t)r * COUT + nt * 16 + m] = acc[nt][j];
    }
  }
}

extern "C" void kernel_launch(void* const* d_in, const int* in_sizes, int n_in,
                              void* d_out, int out_size, void* d_ws, size_t ws_size,
                              hipStream_t stream) {
  const float* features = (const float*)d_in[0];
  const float* kern     = (const float*)d_in[1];
  const int*   coords   = (const int*)d_in[2];
  const int*   offsets  = (const int*)d_in[3];
  float* out = (float*)d_out;
  const int n    = in_sizes[0] / CIN;
  const int noff = in_sizes[3] / 3;

  char* w = (char*)d_ws;
  auto carve = [&](size_t bytes) -> void* {
    void* p = (void*)w;
    w += (bytes + 255) & ~(size_t)255;
    return p;
  };
  int* keys     = (int*)carve((size_t)n * 4);
  int* skeys    = (int*)carve((size_t)n * 4);
  int* sidx     = (int*)carve((size_t)n * 4);
  int* hist     = (int*)carve((size_t)NBUCK * 4);
  int* counters = (int*)carve((size_t)NBUCK * 4);
  int* starts   = (int*)carve((size_t)(NBUCK + 1) * 4);
  int* okeys    = (int*)carve((size_t)noff * 4);
  _Float16* wt    = (_Float16*)carve((size_t)noff * CIN * COUT * 2);
  _Float16* sfeat = (_Float16*)carve((size_t)n * CIN * 2);

  k_init<<<(NBUCK + 255) / 256, 256, 0, stream>>>(hist, counters);
  k_keys<<<(n + 255) / 256, 256, 0, stream>>>(coords, keys, hist, n);
  k_scan<<<1, 1024, 0, stream>>>(hist, starts);
  k_scatter<<<(n + 255) / 256, 256, 0, stream>>>(keys, starts, counters, skeys, sidx, n);
  k_bsort<<<(NBUCK + 255) / 256, 256, 0, stream>>>(starts, skeys, sidx);
  k_gather<<<(n * 32 + 255) / 256, 256, 0, stream>>>(features, sidx, sfeat, n);
  k_prep_w<<<(noff * 4096 + 255) / 256, 256, 0, stream>>>(kern, wt, noff * 4096);
  k_okeys<<<1, 32, 0, stream>>>(offsets, okeys, noff);
  k_conv<<<(n + 15) / 16, 32, 0, stream>>>(skeys, starts, sfeat, wt, okeys, out, n, noff);
}
